// Graph_sampling_network_67259187855841
// MI455X (gfx1250) — matrix operand order repaired and checked by reading, not compiled, not run on hardware
//
#include <hip/hip_runtime.h>
#include <hip/hip_bf16.h>
#include <hip/hip_fp16.h>

typedef __attribute__((ext_vector_type(16))) _Float16 v16h;
typedef __attribute__((ext_vector_type(8)))  _Float16 v8h;
typedef __attribute__((ext_vector_type(8)))  float    v8f;
typedef __attribute__((ext_vector_type(4)))  unsigned int u32x4;
typedef __attribute__((ext_vector_type(8)))  unsigned int u32x8;

#define N_EDGES 8192
#define DIM     128
#define KDIM    512
#define BATCH   128
#define NSEG    (BATCH + 1)
#define CUTOFF  4096

// ---- workspace layout (bytes, all offsets 256-aligned; total ~6.6 MB) ----
#define WS_W1T    (size_t)0        // 512*128 f16  = 131072
#define WS_CENTER (size_t)131072   // 128*128 f32  = 65536
#define WS_H1     (size_t)196608   // 8192*128 f32 = 4194304
#define WS_XN     (size_t)4390912  // 8192*128 f16 = 2097152
#define WS_DIAG   (size_t)6488064  // 8192 f32
#define WS_RELDOT (size_t)6520832  // 8192 f32
#define WS_SEGMAX (size_t)6553600  // 129 f32 (padded 1024)
#define WS_SEGDEN (size_t)6554624  // 129 f32 (padded 1024)
#define WS_RED    (size_t)6555648  // 8192 f32

static __device__ __forceinline__ v8f wmma_f16(v16h a, v16h b, v8f c) {
    return __builtin_amdgcn_wmma_f32_16x16x32_f16(false, a, false, b,
                                                  (short)0, c, false, false);
}

// TDM: async-copy one contiguous 16KB block (2048 x 8B elements) global -> LDS.
// D# per CDNA5 ISA ch.8: group0 = {ctl(count=1), lds_addr, gaddr_lo,
// gaddr[56:32]|type=2}; group1 = 1-D descriptor, data_size=8B,
// tile_dim0 = tensor_dim0 = tensor_dim0_stride = 2048.
static __device__ __forceinline__ void tdm_load_16k(const _Float16* src,
                                                    unsigned int lds_off) {
    unsigned long long ga = (unsigned long long)(uintptr_t)src;
    u32x4 g0;
    g0[0] = 1u;                                               // count=1, user mode
    g0[1] = lds_off;                                          // lds_addr (bytes)
    g0[2] = (unsigned int)ga;                                 // global_addr[31:0]
    g0[3] = (unsigned int)((ga >> 32) & 0x01FFFFFFu) | 0x80000000u; // [56:32]|type=2
    u32x8 g1;
    g1[0] = 0x00030000u;   // workgroup_mask=0, data_size=3 (8B), no pad/iterate
    g1[1] = 0x08000000u;   // atomic_barrier=0 | tensor_dim0.lo16 (2048) << 16
    g1[2] = 0x00010000u;   // tensor_dim0.hi16=0 | tensor_dim1.lo16 (1) << 16
    g1[3] = 0x08000000u;   // tensor_dim1.hi16=0 | tile_dim0 (2048) << 16
    g1[4] = 0x00000001u;   // tile_dim1=1 | tile_dim2=0
    g1[5] = 2048u;         // tensor_dim0_stride[31:0]
    g1[6] = 0u;            // stride hi / tensor_dim1_stride lo
    g1[7] = 0u;
    asm volatile("tensor_load_to_lds %0, %1" :: "s"(g0), "s"(g1) : "memory");
}

// K0: W1 (512x128 f32, K-major) -> W1T (128x512 f16, N-major); gather center feats.
__global__ __launch_bounds__(256) void k0_prep(const float* __restrict__ W1,
                                               const float* __restrict__ emb,
                                               const int*   __restrict__ cidx,
                                               _Float16* __restrict__ w1t,
                                               float* __restrict__ center) {
    int tid = blockIdx.x * 256 + threadIdx.x;
    int stride = gridDim.x * 256;
    for (int i = tid; i < KDIM * DIM; i += stride) {
        int n = i / KDIM, k = i - n * KDIM;
        w1t[(size_t)n * KDIM + k] = (_Float16)W1[(size_t)k * DIM + n];
    }
    for (int i = tid; i < BATCH * DIM; i += stride) {
        int b = i >> 7, c = i & 127;
        center[i] = emb[(size_t)cidx[b] * DIM + c];
    }
}

// K1: h1 = relu(concat(emb[i],emb[j],efeat,tenc) @ W1 + b1)  -- WMMA f16, K=512
__global__ __launch_bounds__(256) void k1_gemm1(const float* __restrict__ emb,
                                                const int*   __restrict__ eidx,
                                                const float* __restrict__ efeat,
                                                const float* __restrict__ tenc,
                                                const _Float16* __restrict__ w1t,
                                                const float* __restrict__ b1,
                                                float* __restrict__ h1) {
    __shared__ __align__(32) _Float16 hs[16 * KDIM];   // 16 KB staged A tile (f16)
    const int m0 = blockIdx.x * 16;
    for (int idx = threadIdx.x; idx < 16 * KDIM; idx += 256) {
        int r = idx >> 9, k = idx & 511;
        int e = m0 + r;
        float v;
        if (k < 128)      v = emb[(size_t)eidx[e * 2]     * DIM + k];
        else if (k < 256) v = emb[(size_t)eidx[e * 2 + 1] * DIM + (k - 128)];
        else if (k < 384) v = efeat[(size_t)e * DIM + (k - 256)];
        else              v = tenc [(size_t)e * DIM + (k - 384)];
        hs[idx] = (_Float16)v;
    }
    __syncthreads();

    const int lane = threadIdx.x & 31;
    const int wave = threadIdx.x >> 5;
    const int n0   = wave * 16;
    const int row  = lane & 15;
    const int hi   = lane >> 4;       // 0: low half, 1: high half of wave

    const _Float16* ap = hs + row * KDIM + hi * 16;                   // A: row = lane&15
    const _Float16* bp = w1t + (size_t)(n0 + row) * KDIM + hi * 16;   // B: col = lane&15

    v8f c = {};
    for (int ks = 0; ks < 16; ++ks) {
        v8h alo = *(const v8h*)(ap + ks * 32);
        v8h ahi = *(const v8h*)(ap + ks * 32 + 8);
        v16h a = __builtin_shufflevector(alo, ahi, 0,1,2,3,4,5,6,7,8,9,10,11,12,13,14,15);
        v16h b = *(const v16h*)(bp + ks * 32);
        c = wmma_f16(a, b, c);
    }
    const int n = n0 + row;
    const float bias = b1[n];
    for (int r = 0; r < 8; ++r) {
        int m = m0 + r + hi * 8;                     // C/D: lanes 0-15 -> M=r, 16-31 -> M=r+8
        float v = c[r] + bias;
        h1[(size_t)m * DIM + n] = v > 0.f ? v : 0.f;
    }
}

// K2: per-row L2 norm -> xn (f16), diag = ||xn_f16||^2, reldot = h1 . center[batch-1]
__global__ __launch_bounds__(256) void k2_norm(const float* __restrict__ h1,
                                               const float* __restrict__ center,
                                               const int*   __restrict__ batch,
                                               _Float16* __restrict__ xn,
                                               float* __restrict__ diag,
                                               float* __restrict__ reldot) {
    const int lane = threadIdx.x & 31;
    const int wave = threadIdx.x >> 5;
    const int row  = blockIdx.x * 8 + wave;
    const float* hr = h1 + (size_t)row * DIM;
    float v[4], ss = 0.f;
    for (int q = 0; q < 4; ++q) { v[q] = hr[lane + q * 32]; ss += v[q] * v[q]; }
    for (int off = 16; off; off >>= 1) ss += __shfl_xor(ss, off, 32);
    float nrm = sqrtf(ss); nrm = nrm > 1e-12f ? nrm : 1e-12f;
    const float inv = 1.f / nrm;
    const float* cr = center + (size_t)(batch[row] - 1) * DIM;
    float dd = 0.f, rd = 0.f;
    for (int q = 0; q < 4; ++q) {
        _Float16 xh = (_Float16)(v[q] * inv);
        xn[(size_t)row * DIM + lane + q * 32] = xh;
        float xf = (float)xh;
        dd += xf * xf;
        rd += v[q] * cr[lane + q * 32];
    }
    for (int off = 16; off; off >>= 1) {
        dd += __shfl_xor(dd, off, 32);
        rd += __shfl_xor(rd, off, 32);
    }
    if (lane == 0) { diag[row] = dd; reldot[row] = rd; }
}

// K3: segment max + sum(exp) over 129 segments (one block per segment)
__global__ __launch_bounds__(256) void k3_seg(const float* __restrict__ reldot,
                                              const int*   __restrict__ batch,
                                              float* __restrict__ segmax,
                                              float* __restrict__ segden) {
    __shared__ float red[256];
    const int seg = blockIdx.x;
    float m = -3.4e38f;
    for (int e = threadIdx.x; e < N_EDGES; e += 256)
        if (batch[e] == seg) { float d = reldot[e]; m = d > m ? d : m; }
    red[threadIdx.x] = m; __syncthreads();
    for (int s = 128; s; s >>= 1) {
        if (threadIdx.x < s) { float o = red[threadIdx.x + s]; if (o > red[threadIdx.x]) red[threadIdx.x] = o; }
        __syncthreads();
    }
    const float sm = red[0]; __syncthreads();
    float a = 0.f;
    for (int e = threadIdx.x; e < N_EDGES; e += 256)
        if (batch[e] == seg) a += __expf(reldot[e] - sm);
    red[threadIdx.x] = a; __syncthreads();
    for (int s = 128; s; s >>= 1) {
        if (threadIdx.x < s) red[threadIdx.x] += red[threadIdx.x + s];
        __syncthreads();
    }
    if (threadIdx.x == 0) { segmax[seg] = sm; segden[seg] = red[0]; }
}

// K4: redundancy = 1 - softmax-diag of xn @ xn^T (streaming; shift fixed at 1
// since rows are unit vectors => dots<=1 and diag==max). WMMA f16, K=128.
// One i-tile per wave (8 per block); B streamed through LDS in 16KB chunks via
// TDM (tensor_load_to_lds), double-buffered, TENSORcnt-tracked.
#define CH_ROWS 64
#define NCH     (N_EDGES / CH_ROWS)
__global__ __launch_bounds__(256) void k4_redundancy(const _Float16* __restrict__ xn,
                                                     const float* __restrict__ diag,
                                                     float* __restrict__ red) {
    __shared__ __align__(32) _Float16 tile[2][CH_ROWS * DIM];  // 2 x 16 KB

    const int lane = threadIdx.x & 31;
    const int wave = threadIdx.x >> 5;
    const int i0   = (blockIdx.x * 8 + wave) * 16;   // this wave's 16 output rows
    const int row  = lane & 15;
    const int hi   = lane >> 4;

    const unsigned int lds0 = (unsigned int)(uintptr_t)&tile[0][0];
    const unsigned int lds1 = (unsigned int)(uintptr_t)&tile[1][0];

    // A fragments (this wave's 16 rows x K=128), reused for all 512 j tiles
    v16h a[4];
    {
        const _Float16* ap = xn + (size_t)(i0 + row) * DIM + hi * 16;
        for (int ks = 0; ks < 4; ++ks) {
            v8h lo = *(const v8h*)(ap + ks * 32);
            v8h hq = *(const v8h*)(ap + ks * 32 + 8);
            a[ks] = __builtin_shufflevector(lo, hq, 0,1,2,3,4,5,6,7,8,9,10,11,12,13,14,15);
        }
    }

    // prologue: stage chunk 0
    if (wave == 0) {
        tdm_load_16k(xn, lds0);
        __builtin_amdgcn_s_wait_tensorcnt(0);
    }
    __syncthreads();

    float s[8] = {0.f, 0.f, 0.f, 0.f, 0.f, 0.f, 0.f, 0.f};
    for (int ch = 0; ch < NCH; ++ch) {
        const int cur = ch & 1;
        if (wave == 0 && ch + 1 < NCH)   // issue next chunk into the idle buffer
            tdm_load_16k(xn + (size_t)(ch + 1) * CH_ROWS * DIM, cur ? lds0 : lds1);

        const _Float16* tp = tile[cur];
        for (int jl = 0; jl < 4; ++jl) {             // 4 j-tiles per chunk
            const _Float16* bp = tp + (jl * 16 + row) * DIM + hi * 16;
            v8f c = {};
            for (int ks = 0; ks < 4; ++ks) {
                v16h b = *(const v16h*)(bp + ks * 32);
                c = wmma_f16(a[ks], b, c);
            }
            for (int r = 0; r < 8; ++r) s[r] += __expf(c[r] - 1.0f);
        }

        if (wave == 0 && ch + 1 < NCH)
            __builtin_amdgcn_s_wait_tensorcnt(0);
        __syncthreads();
    }

    // reduce across the 16 lanes that share each output row
    for (int r = 0; r < 8; ++r)
        for (int off = 8; off; off >>= 1) s[r] += __shfl_xor(s[r], off, 16);
    if ((lane & 15) == 0) {
        for (int r = 0; r < 8; ++r) {
            int i = i0 + r + hi * 8;
            red[i] = 1.f - __expf(diag[i] - 1.f) / s[r];
        }
    }
}

// K5: log_alpha = [red, rel, h1] @ W3 + b3 ; probs = sigmoid
__global__ __launch_bounds__(256) void k5_score(const float* __restrict__ h1,
                                                const float* __restrict__ W3,
                                                const float* __restrict__ b3,
                                                const float* __restrict__ red,
                                                const float* __restrict__ reldot,
                                                const int*   __restrict__ batch,
                                                const float* __restrict__ segmax,
                                                const float* __restrict__ segden,
                                                float* __restrict__ probs) {
    const int lane = threadIdx.x & 31;
    const int wave = threadIdx.x >> 5;
    const int e    = blockIdx.x * 8 + wave;
    const float* hr = h1 + (size_t)e * DIM;
    float acc = 0.f;
    for (int q = 0; q < 4; ++q) {
        int c = lane + q * 32;
        acc += hr[c] * W3[2 + c];
    }
    for (int off = 16; off; off >>= 1) acc += __shfl_xor(acc, off, 32);
    if (lane == 0) {
        int b = batch[e];
        float rel = __expf(reldot[e] - segmax[b]) / segden[b];
        float la  = acc + red[e] * W3[0] + rel * W3[1] + b3[0];
        probs[e]  = 1.f / (1.f + __expf(-la));
    }
}

// K6: stable-argsort rank via pairwise counting from LDS; mask = rank >= 4096
__global__ __launch_bounds__(256) void k6_rank(const float* __restrict__ probs,
                                               float* __restrict__ mask) {
    __shared__ float lp[N_EDGES];   // 32 KB
    for (int j = threadIdx.x; j < N_EDGES; j += 256) lp[j] = probs[j];
    __syncthreads();
    const int i = blockIdx.x * 256 + threadIdx.x;
    const float p = lp[i];
    int cnt = 0;
    for (int j = 0; j < N_EDGES; ++j) {
        float q = lp[j];
        cnt += (q < p) || (q == p && j < i);
    }
    mask[i] = (cnt >= CUTOFF) ? 1.0f : 0.0f;
}

extern "C" void kernel_launch(void* const* d_in, const int* in_sizes, int n_in,
                              void* d_out, int out_size, void* d_ws, size_t ws_size,
                              hipStream_t stream) {
    const float* emb   = (const float*)d_in[0];
    const int*   eidx  = (const int*)  d_in[1];
    const float* tenc  = (const float*)d_in[2];
    const float* efeat = (const float*)d_in[3];
    const int*   batch = (const int*)  d_in[4];
    const int*   cidx  = (const int*)  d_in[5];
    const float* W1    = (const float*)d_in[6];
    const float* b1    = (const float*)d_in[7];
    const float* W3    = (const float*)d_in[8];
    const float* b3    = (const float*)d_in[9];

    char* ws = (char*)d_ws;
    _Float16* w1t    = (_Float16*)(ws + WS_W1T);
    float*    center = (float*)   (ws + WS_CENTER);
    float*    h1     = (float*)   (ws + WS_H1);
    _Float16* xn     = (_Float16*)(ws + WS_XN);
    float*    diag   = (float*)   (ws + WS_DIAG);
    float*    reldot = (float*)   (ws + WS_RELDOT);
    float*    segmax = (float*)   (ws + WS_SEGMAX);
    float*    segden = (float*)   (ws + WS_SEGDEN);
    float*    red    = (float*)   (ws + WS_RED);

    float* probs = (float*)d_out;            // [0, 8192)
    float* mask  = (float*)d_out + N_EDGES;  // [8192, 16384)

    k0_prep      <<<64,  256, 0, stream>>>(W1, emb, cidx, w1t, center);
    k1_gemm1     <<<N_EDGES / 16, 256, 0, stream>>>(emb, eidx, efeat, tenc, w1t, b1, h1);
    k2_norm      <<<N_EDGES / 8,  256, 0, stream>>>(h1, center, batch, xn, diag, reldot);
    k3_seg       <<<NSEG, 256, 0, stream>>>(reldot, batch, segmax, segden);
    k4_redundancy<<<N_EDGES / (16 * 8), 256, 0, stream>>>(xn, diag, red);
    k5_score     <<<N_EDGES / 8,  256, 0, stream>>>(h1, W3, b3, red, reldot, batch,
                                                    segmax, segden, probs);
    k6_rank      <<<N_EDGES / 256, 256, 0, stream>>>(probs, mask);
}
